// FeatureAggregationBlock_33818572489232
// MI455X (gfx1250) — compile-verified, matrix operand order checked by reading, not run on hardware
//
#include <hip/hip_runtime.h>
#include <hip/hip_bf16.h>

#define DEVINL __device__ __forceinline__

typedef __attribute__((ext_vector_type(16))) _Float16 v16h;
typedef __attribute__((ext_vector_type(8)))  _Float16 v8h;
typedef __attribute__((ext_vector_type(8)))  float    v8f;

static constexpr int N  = 1536;
static constexpr int H  = 8;
static constexpr int C  = 256;   // cls channels
static constexpr int D  = 32;    // cls head dim
static constexpr int CR = 64;    // reg channels
static constexpr int DR = 8;     // reg head dim
static constexpr int NT = N / 16; // 96 tiles per dimension

// ---------------------------------------------------------------------------
// WMMA fragment loaders (wave32, 16-bit A/B layout per CDNA5 ISA 7.12.2):
// lane l: row/col = l&15, K-base = (l>>4)*8; a[i]=K(kb+i), a[8+i]=K(kb+16+i).
// Each half-fragment is 8 contiguous f16 = 16 B -> one global_load_b128.
// For S = Q @ K^T the B fragment reads the K matrix row-major as transposed,
// which is the same addressing pattern as the A fragment.
// ---------------------------------------------------------------------------
DEVINL v16h ld_frag(const _Float16* __restrict__ base, int r0, int ld, int k0) {
  const int lane = threadIdx.x & 31;
  const int sub  = lane & 15;
  const int grp  = lane >> 4;
  const _Float16* p = base + (size_t)(r0 + sub) * ld + k0 + grp * 8;
  const v8h lo = *(const v8h*)p;        // 16B aligned by construction
  const v8h hi = *(const v8h*)(p + 16);
  v16h a;
#pragma unroll
  for (int i = 0; i < 8; ++i) { a[i] = lo[i]; a[8 + i] = hi[i]; }
  return a;
}

// 8 valid K-elements (reg branch head slice of a [n][64] array), zero-padded
// to the K=32 fragment. h*DR is 16B aligned (h*8 f16).
DEVINL v16h ld_frag_pad8(const _Float16* __restrict__ base, int r0, int h) {
  const int lane = threadIdx.x & 31;
  const int sub  = lane & 15;
  const int grp  = lane >> 4;
  v16h a;
#pragma unroll
  for (int i = 0; i < 16; ++i) a[i] = (_Float16)0.0f;
  if (grp == 0) {
    const v8h lo = *(const v8h*)(base + (size_t)(r0 + sub) * CR + h * DR);
#pragma unroll
    for (int i = 0; i < 8; ++i) a[i] = lo[i];  // K=0..7 live, K=16..23 zero
  }
  return a;
}

DEVINL v8f wmma16(v16h a, v16h b, v8f c) {
  return __builtin_amdgcn_wmma_f32_16x16x32_f16(false, a, false, b, (short)0, c,
                                                false, false);
}

// ---------------------------------------------------------------------------
// Zero helper (osum is atomically accumulated -> re-zero every launch so the
// kernel stays deterministic under graph replay).
// ---------------------------------------------------------------------------
__global__ void k_zero(float* __restrict__ p, int n) {
  int t = blockIdx.x * blockDim.x + threadIdx.x;
  if (t < n) p[t] = 0.0f;
}

// rosum[i] = 1 / osum[i]  (diagonal of ObjU is always 1, so osum > 0)
__global__ void k_rcp(const float* __restrict__ osum, float* __restrict__ rosum,
                      int n) {
  int t = blockIdx.x * blockDim.x + threadIdx.x;
  if (t < n) rosum[t] = 1.0f / osum[t];
}

// ---------------------------------------------------------------------------
// Generic WMMA GEMM: O[M,Nc] = X[M,K] @ W[K,Nc], f32 in/out, f16 compute.
// One wave per 16x16 output tile, 4 waves per block along Nc.
// ---------------------------------------------------------------------------
__global__ void __launch_bounds__(128)
k_gemm_wmma(const float* __restrict__ X, const float* __restrict__ W,
            float* __restrict__ O, int M, int K, int Nc) {
  const int wave = threadIdx.x >> 5;
  const int nt   = blockIdx.x * 4 + wave;
  const int mt   = blockIdx.y;
  if (nt * 16 >= Nc || mt * 16 >= M) return;
  const int lane = threadIdx.x & 31;
  const int sub  = lane & 15;
  const int grp  = lane >> 4;

  v8f c = {};
  for (int k0 = 0; k0 < K; k0 += 32) {
    const int kb = k0 + grp * 8;
    v16h a, b;
    // A operand: two float4 loads (32B-aligned), convert to f16
    const float4* xp = (const float4*)(X + (size_t)(mt * 16 + sub) * K + kb);
    const float4 x0 = xp[0], x1 = xp[1];
    const float4* xq = (const float4*)(X + (size_t)(mt * 16 + sub) * K + kb + 16);
    const float4 x2 = xq[0], x3 = xq[1];
    a[0] = (_Float16)x0.x; a[1] = (_Float16)x0.y; a[2] = (_Float16)x0.z; a[3] = (_Float16)x0.w;
    a[4] = (_Float16)x1.x; a[5] = (_Float16)x1.y; a[6] = (_Float16)x1.z; a[7] = (_Float16)x1.w;
    a[8]  = (_Float16)x2.x; a[9]  = (_Float16)x2.y; a[10] = (_Float16)x2.z; a[11] = (_Float16)x2.w;
    a[12] = (_Float16)x3.x; a[13] = (_Float16)x3.y; a[14] = (_Float16)x3.z; a[15] = (_Float16)x3.w;
    // B operand: column-strided, scalar loads
    const float* wp = W + (size_t)kb * Nc + nt * 16 + sub;
#pragma unroll
    for (int i = 0; i < 8; ++i) b[i] = (_Float16)wp[(size_t)i * Nc];
#pragma unroll
    for (int i = 0; i < 8; ++i) b[8 + i] = (_Float16)wp[(size_t)(16 + i) * Nc];
    c = wmma16(a, b, c);
  }
  float* op = O + (size_t)(mt * 16 + grp * 8) * Nc + nt * 16 + sub;
#pragma unroll
  for (int m = 0; m < 8; ++m) op[(size_t)m * Nc] = c[m];
}

// ---------------------------------------------------------------------------
// L2-normalize + pack (cls branch). qkv layout: [n][3][H][D].
// Outputs: Qn/Kn = [h][n][D] f16 (normalized), VnCat = [n][C] f16 (normalized,
// head-concat), Vf = [n][C] f32 (raw v, head-concat).
// ---------------------------------------------------------------------------
__global__ void k_norm_cls(const float* __restrict__ qkv,
                           _Float16* __restrict__ Qn, _Float16* __restrict__ Kn,
                           _Float16* __restrict__ VnCat, float* __restrict__ Vf) {
  int t = blockIdx.x * blockDim.x + threadIdx.x;
  if (t >= N * H) return;
  const int n = t >> 3, h = t & 7;
#pragma unroll
  for (int s = 0; s < 3; ++s) {
    const float* x = qkv + (size_t)n * (3 * C) + s * C + h * D;
    float ss = 0.0f;
#pragma unroll
    for (int d = 0; d < D; ++d) ss += x[d] * x[d];
    const float rn = rsqrtf(ss);
    if (s == 0) {
      _Float16* q = Qn + (size_t)h * N * D + (size_t)n * D;
#pragma unroll
      for (int d = 0; d < D; ++d) q[d] = (_Float16)(x[d] * rn);
    } else if (s == 1) {
      _Float16* k = Kn + (size_t)h * N * D + (size_t)n * D;
#pragma unroll
      for (int d = 0; d < D; ++d) k[d] = (_Float16)(x[d] * rn);
    } else {
#pragma unroll
      for (int d = 0; d < D; ++d) {
        VnCat[(size_t)n * C + h * D + d] = (_Float16)(x[d] * rn);
        Vf[(size_t)n * C + h * D + d]    = x[d];
      }
    }
  }
}

// Reg branch. qkv layout [n][3][H][DR]. Qrn/Krn/Vrn head-concat [n][CR] f16.
__global__ void k_norm_reg(const float* __restrict__ qkv,
                           _Float16* __restrict__ Qrn, _Float16* __restrict__ Krn,
                           _Float16* __restrict__ VrnCat, float* __restrict__ Vrf) {
  int t = blockIdx.x * blockDim.x + threadIdx.x;
  if (t >= N * H) return;
  const int n = t >> 3, h = t & 7;
#pragma unroll
  for (int s = 0; s < 3; ++s) {
    const float* x = qkv + (size_t)n * (3 * CR) + s * CR + h * DR;
    float ss = 0.0f;
#pragma unroll
    for (int d = 0; d < DR; ++d) ss += x[d] * x[d];
    const float rn = rsqrtf(ss);
    _Float16* dst = (s == 0) ? Qrn : (s == 1) ? Krn : VrnCat;
#pragma unroll
    for (int d = 0; d < DR; ++d) {
      dst[(size_t)n * CR + h * DR + d] = (_Float16)(x[d] * rn);
      if (s == 2) Vrf[(size_t)n * CR + h * DR + d] = x[d];
    }
  }
}

// ---------------------------------------------------------------------------
// Mask kernel: per 16x16 tile compute
//   simraw = mean_h(vn_h . vn_h) == (VnCat_i . VnCat_j)/8   (K=256 WMMA)
//   objraw = mean_h(vrn_h . vrn_h) == (VrnCat_i . VrnCat_j)/8 (K=64 WMMA)
//   sim  = (simraw > .75) * clsmask * fgmask -> written to all 8 head slices
//   objU = (objraw > .75) * sim             -> workspace + row sums (osum)
// ---------------------------------------------------------------------------
__global__ void __launch_bounds__(256)
k_masks(const _Float16* __restrict__ VnCat, const _Float16* __restrict__ VrnCat,
        const float* __restrict__ cls_score, const float* __restrict__ fg_score,
        float* __restrict__ sim_out, float* __restrict__ objU,
        float* __restrict__ osum) {
  const int wave = threadIdx.x >> 5;
  const int jt = blockIdx.x * 8 + wave;
  const int it = blockIdx.y;
  const int i0 = it * 16, j0 = jt * 16;
  const int lane = threadIdx.x & 31;
  const int sub  = lane & 15;
  const int grp  = lane >> 4;

  v8f cs = {};
  for (int k0 = 0; k0 < C; k0 += 32) {
    v16h a = ld_frag(VnCat, i0, C, k0);
    v16h b = ld_frag(VnCat, j0, C, k0);  // B^T addressing == A addressing
    cs = wmma16(a, b, cs);
  }
  v8f co = {};
  for (int k0 = 0; k0 < CR; k0 += 32) {
    v16h a = ld_frag(VrnCat, i0, CR, k0);
    v16h b = ld_frag(VrnCat, j0, CR, k0);
    co = wmma16(a, b, co);
  }

  const float csj = cls_score[j0 + sub];
  const float fgj = fg_score[j0 + sub];
  float rp[8];
#pragma unroll
  for (int m = 0; m < 8; ++m) {
    const int r = i0 + grp * 8 + m;
    const float cm = (csj > cls_score[r] - 0.1f) ? 1.0f : 0.0f;
    const float fm = (fgj > fg_score[r] - 0.1f) ? 1.0f : 0.0f;
    const float sm = ((cs[m] * 0.125f) > 0.75f) ? (cm * fm) : 0.0f;
    const float ou = ((co[m] * 0.125f) > 0.75f) ? sm : 0.0f;
    const size_t idx = (size_t)r * N + j0 + sub;
#pragma unroll
    for (int h = 0; h < H; ++h) sim_out[(size_t)h * N * N + idx] = sm;
    objU[idx] = ou;
    rp[m] = ou;
  }
  // reduce row partials across the 16 lanes holding the same rows
#pragma unroll
  for (int off = 8; off >= 1; off >>= 1)
#pragma unroll
    for (int m = 0; m < 8; ++m) rp[m] += __shfl_xor(rp[m], off, 16);
  if (sub == 0) {
#pragma unroll
    for (int m = 0; m < 8; ++m) atomicAdd(&osum[i0 + grp * 8 + m], rp[m]);
  }
}

// ---------------------------------------------------------------------------
// obj_mask output: obj_out[h,i,j] = ObjU[i,j] * rosum[i], replicated over the
// 8 head slices. This is the single largest output (75 MB) so it gets its own
// massively parallel streaming kernel (9216 blocks) to saturate HBM stores.
// ---------------------------------------------------------------------------
__global__ void k_objnorm(const float* __restrict__ objU,
                          const float* __restrict__ rosum,
                          float* __restrict__ obj_out) {
  const size_t t = (size_t)blockIdx.x * blockDim.x + threadIdx.x;
  if (t >= (size_t)N * N) return;
  const int i = (int)(t / N);
  const float on = objU[t] * rosum[i];
#pragma unroll
  for (int h = 0; h < H; ++h) obj_out[(size_t)h * N * N + t] = on;
}

// ---------------------------------------------------------------------------
// Attention kernel: one block per 16-row i-tile, 8 waves = 8 heads.
// Pass 1: online-softmax row stats per head (WMMA logits + lane-local running
//         max/sum, exact 16-lane shuffle merge).
// Pass 2: recompute logits, apply softmax + masks + obj normalization, route
//         attn tile through LDS, f32 PV accumulation.
// ---------------------------------------------------------------------------
__global__ void __launch_bounds__(256)
k_attn(const _Float16* __restrict__ Qn, const _Float16* __restrict__ Kn,
       const _Float16* __restrict__ Qrn, const _Float16* __restrict__ Krn,
       const float* __restrict__ Vf, const float* __restrict__ Vrf,
       const float* __restrict__ cls_score,
       const float* __restrict__ objU, const float* __restrict__ rosum,
       const float* __restrict__ sim0,
       float* __restrict__ out_cls, float* __restrict__ out_reg) {
  __shared__ float lds_p[8][16][17];
  const int wave = threadIdx.x >> 5;
  const int h    = wave;
  const int i0   = blockIdx.x * 16;
  const int lane = threadIdx.x & 31;
  const int sub  = lane & 15;
  const int grp  = lane >> 4;
  const float scale = 0.17677669529663687f;  // (C/H)^-0.5 = 1/sqrt(32)
  const _Float16* KnH = Kn + (size_t)h * N * D;

  const v16h aq = ld_frag(Qn + (size_t)h * N * D, i0, D, 0);
  const v16h ar = ld_frag_pad8(Qrn, i0, h);

  float csi[8];
#pragma unroll
  for (int m = 0; m < 8; ++m) csi[m] = cls_score[i0 + grp * 8 + m];

  float mC[8], sC[8], mR[8], sR[8];
#pragma unroll
  for (int m = 0; m < 8; ++m) {
    mC[m] = -3.0e38f; sC[m] = 0.0f; mR[m] = -3.0e38f; sR[m] = 0.0f;
  }

  // ---- pass 1: row stats ----
  for (int jt = 0; jt < NT; ++jt) {
    const int j0 = jt * 16;
    if (jt + 1 < NT) {  // gfx1250 global_prefetch of the next K tile
      __builtin_prefetch(KnH + (size_t)(j0 + 16 + sub) * D, 0, 1);
    }
    v16h bk = ld_frag(KnH, j0, D, 0);
    v8f c = {};
    c = wmma16(aq, bk, c);
    v16h br = ld_frag_pad8(Krn, j0, h);
    v8f cr = {};
    cr = wmma16(ar, br, cr);
    const float csj = cls_score[j0 + sub];
#pragma unroll
    for (int m = 0; m < 8; ++m) {
      const float cm = (csj > csi[m] - 0.1f) ? 1.0f : 0.0f;
      const float lc = c[m] * scale * csj * cm;   // masked logit is 0 (mult.)
      float nm = fmaxf(mC[m], lc);
      sC[m] = sC[m] * __expf(mC[m] - nm) + __expf(lc - nm);
      mC[m] = nm;
      const float lr = cr[m] * scale;
      nm = fmaxf(mR[m], lr);
      sR[m] = sR[m] * __expf(mR[m] - nm) + __expf(lr - nm);
      mR[m] = nm;
    }
  }
  // exact cross-lane softmax merge within each 16-lane group
#pragma unroll
  for (int off = 8; off >= 1; off >>= 1) {
#pragma unroll
    for (int m = 0; m < 8; ++m) {
      float om = __shfl_xor(mC[m], off, 16);
      float os = __shfl_xor(sC[m], off, 16);
      float nm = fmaxf(mC[m], om);
      sC[m] = sC[m] * __expf(mC[m] - nm) + os * __expf(om - nm);
      mC[m] = nm;
      om = __shfl_xor(mR[m], off, 16);
      os = __shfl_xor(sR[m], off, 16);
      nm = fmaxf(mR[m], om);
      sR[m] = sR[m] * __expf(mR[m] - nm) + os * __expf(om - nm);
      mR[m] = nm;
    }
  }
  float rC[8], rR[8], ros[8];
#pragma unroll
  for (int m = 0; m < 8; ++m) {
    rC[m]  = 1.0f / sC[m];
    rR[m]  = 1.0f / sR[m];
    ros[m] = rosum[i0 + grp * 8 + m];
  }

  float accC[16], accR[4];
#pragma unroll
  for (int t = 0; t < 16; ++t) accC[t] = 0.0f;
#pragma unroll
  for (int t = 0; t < 4; ++t) accR[t] = 0.0f;

  // ---- pass 2: outputs ----
  for (int jt = 0; jt < NT; ++jt) {
    const int j0 = jt * 16;
    if (jt + 1 < NT) {
      __builtin_prefetch(Vf + (size_t)(j0 + 16 + sub) * C + h * D, 0, 1);
    }
    v16h bk = ld_frag(KnH, j0, D, 0);
    v8f c = {};
    c = wmma16(aq, bk, c);
    v16h br = ld_frag_pad8(Krn, j0, h);
    v8f cr = {};
    cr = wmma16(ar, br, cr);
    const float csj = cls_score[j0 + sub];

    __syncthreads();  // protect LDS vs previous iteration's reads
#pragma unroll
    for (int m = 0; m < 8; ++m) {
      const float cm = (csj > csi[m] - 0.1f) ? 1.0f : 0.0f;
      const float lc = c[m] * scale * csj * cm;
      const float lr = cr[m] * scale;
      const float pc = __expf(lc - mC[m]) * rC[m];
      const float pr = __expf(lr - mR[m]) * rR[m];
      const int r = i0 + grp * 8 + m;
      const size_t idx = (size_t)r * N + j0 + sub;
      const float sm = sim0[idx];
      const float on = objU[idx] * ros[m];
      lds_p[wave][grp * 8 + m][sub] = 0.5f * (pc + pr) * sm * on;
    }
    __syncthreads();

    // PV accumulation: lane owns row=sub, 16 (cls) / 4 (reg) contiguous d's
#pragma unroll
    for (int jl = 0; jl < 16; ++jl) {
      const float p = lds_p[wave][sub][jl];
      const float4* vp = (const float4*)(Vf + (size_t)(j0 + jl) * C + h * D + grp * 16);
      const float4 v0 = vp[0], v1 = vp[1], v2 = vp[2], v3 = vp[3];
      accC[0]  += p * v0.x; accC[1]  += p * v0.y; accC[2]  += p * v0.z; accC[3]  += p * v0.w;
      accC[4]  += p * v1.x; accC[5]  += p * v1.y; accC[6]  += p * v1.z; accC[7]  += p * v1.w;
      accC[8]  += p * v2.x; accC[9]  += p * v2.y; accC[10] += p * v2.z; accC[11] += p * v2.w;
      accC[12] += p * v3.x; accC[13] += p * v3.y; accC[14] += p * v3.z; accC[15] += p * v3.w;
      const float4 w0 = *(const float4*)(Vrf + (size_t)(j0 + jl) * CR + h * DR + grp * 4);
      accR[0] += p * w0.x; accR[1] += p * w0.y; accR[2] += p * w0.z; accR[3] += p * w0.w;
    }
  }

  const int row = i0 + sub;
  float* oc = out_cls + (size_t)row * (2 * C) + h * D + grp * 16;
#pragma unroll
  for (int t = 0; t < 16; ++t) oc[t] = accC[t];
  float* og = out_reg + (size_t)row * (2 * CR) + h * DR + grp * 4;
#pragma unroll
  for (int t = 0; t < 4; ++t) og[t] = accR[t];
}

// ---------------------------------------------------------------------------
// Concatenate the residual halves: out_cls[:,C:2C]=x_cls, out_reg[:,CR:2CR]=x_reg
// ---------------------------------------------------------------------------
__global__ void k_concat(const float* __restrict__ x_cls,
                         const float* __restrict__ x_reg,
                         float* __restrict__ out_cls, float* __restrict__ out_reg) {
  int t = blockIdx.x * blockDim.x + threadIdx.x;
  const int n1 = N * C;
  if (t < n1) {
    const int n = t / C, c = t % C;
    out_cls[(size_t)n * (2 * C) + C + c] = x_cls[t];
  } else {
    const int t2 = t - n1;
    if (t2 < N * CR) {
      const int n = t2 / CR, c = t2 % CR;
      out_reg[(size_t)n * (2 * CR) + CR + c] = x_reg[t2];
    }
  }
}

// ---------------------------------------------------------------------------
extern "C" void kernel_launch(void* const* d_in, const int* in_sizes, int n_in,
                              void* d_out, int out_size, void* d_ws, size_t ws_size,
                              hipStream_t stream) {
  const float* x_cls     = (const float*)d_in[0];  // [1,N,C]
  const float* x_reg     = (const float*)d_in[1];  // [1,N,CR]
  const float* cls_score = (const float*)d_in[2];  // [N]
  const float* fg_score  = (const float*)d_in[3];  // [1,1,1,N]
  const float* W_cls     = (const float*)d_in[4];  // [C,3C]
  const float* W_reg     = (const float*)d_in[5];  // [CR,3CR]

  // d_out: out_cls [N,2C] | out_reg [N,2CR] | sim [H,N,N] | obj [H,N,N]
  float* out_cls = (float*)d_out;
  float* out_reg = out_cls + (size_t)N * 2 * C;
  float* sim_out = out_reg + (size_t)N * 2 * CR;
  float* obj_out = sim_out + (size_t)H * N * N;

  // workspace carve (all offsets 256B aligned)
  char* ws = (char*)d_ws;
  size_t off = 0;
  auto carve = [&](size_t bytes) -> void* {
    void* p = ws + off;
    off = (off + bytes + 255) & ~((size_t)255);
    return p;
  };
  float*    qkvc  = (float*)carve((size_t)N * 3 * C * sizeof(float));
  float*    qkvr  = (float*)carve((size_t)N * 3 * CR * sizeof(float));
  _Float16* Qn    = (_Float16*)carve((size_t)H * N * D * sizeof(_Float16));
  _Float16* Kn    = (_Float16*)carve((size_t)H * N * D * sizeof(_Float16));
  _Float16* VnCat = (_Float16*)carve((size_t)N * C * sizeof(_Float16));
  float*    Vf    = (float*)carve((size_t)N * C * sizeof(float));
  _Float16* Qrn   = (_Float16*)carve((size_t)N * CR * sizeof(_Float16));
  _Float16* Krn   = (_Float16*)carve((size_t)N * CR * sizeof(_Float16));
  _Float16* Vrn   = (_Float16*)carve((size_t)N * CR * sizeof(_Float16));
  float*    Vrf   = (float*)carve((size_t)N * CR * sizeof(float));
  float*    ObjU  = (float*)carve((size_t)N * N * sizeof(float));
  float*    osum  = (float*)carve((size_t)N * sizeof(float));
  float*    rosum = (float*)carve((size_t)N * sizeof(float));

  // osum must be re-zeroed every call (accumulated by atomics)
  k_zero<<<(N + 255) / 256, 256, 0, stream>>>(osum, N);

  // QKV projections (WMMA)
  k_gemm_wmma<<<dim3((3 * C) / 64, NT), 128, 0, stream>>>(x_cls, W_cls, qkvc,
                                                          N, C, 3 * C);
  k_gemm_wmma<<<dim3((3 * CR) / 64, NT), 128, 0, stream>>>(x_reg, W_reg, qkvr,
                                                           N, CR, 3 * CR);

  // normalize + pack
  k_norm_cls<<<(N * H + 255) / 256, 256, 0, stream>>>(qkvc, Qn, Kn, VnCat, Vf);
  k_norm_reg<<<(N * H + 255) / 256, 256, 0, stream>>>(qkvr, Qrn, Krn, Vrn, Vrf);

  // similarity masks (WMMA K=256 / K=64) + obj row sums
  k_masks<<<dim3(NT / 8, NT), 256, 0, stream>>>(VnCat, Vrn, cls_score, fg_score,
                                                sim_out, ObjU, osum);

  // obj row-sum reciprocals, then the 75MB obj_mask output with max parallelism
  k_rcp<<<(N + 255) / 256, 256, 0, stream>>>(osum, rosum, N);
  k_objnorm<<<((size_t)N * N + 255) / 256, 256, 0, stream>>>(ObjU, rosum, obj_out);

  // fused dual-branch attention (WMMA logits, online softmax, PV)
  k_attn<<<NT, 256, 0, stream>>>(Qn, Kn, Qrn, Krn, Vf, Vrf, cls_score, ObjU,
                                 rosum, sim_out, out_cls, out_reg);

  // residual concatenation
  k_concat<<<(N * C + N * CR + 255) / 256, 256, 0, stream>>>(x_cls, x_reg,
                                                             out_cls, out_reg);
}
